// SelfAttentionlayer_16844861735386
// MI455X (gfx1250) — matrix operand order repaired and checked by reading, not compiled, not run on hardware
//
#include <hip/hip_runtime.h>

typedef __attribute__((ext_vector_type(16))) _Float16 v16h;
typedef __attribute__((ext_vector_type(8)))  _Float16 v8h;
typedef __attribute__((ext_vector_type(8)))  float    v8f;

#define AMP    257
#define ATT    257
#define PAD    288          // 9 * 32, padded feature dim
#define KSTEPS 9            // PAD / 32
#define NT     18           // PAD / 16
#define BATCH  8
#define TSEQ   2048
#define ROWS   (BATCH*TSEQ) // 16384
#define RTILES (ROWS/16)    // 1024

__device__ __forceinline__ v8f wmma_f16(v16h a, v16h b, v8f c) {
  // 8 args: (neg_a, A, neg_b, B, c_mod, C, reuse_a, reuse_b)
  return __builtin_amdgcn_wmma_f32_16x16x32_f16(false, a, false, b, (short)0, c, false, false);
}

// A fragment (16x32 f16, M=row striped per lane) from f16 row-major matrix.
// lanes 0-15: elems 0..15 = K kb..kb+15
// lanes 16-31: elems 0..15 = K kb+16..kb+31.
__device__ __forceinline__ v16h load_a16(const _Float16* p, int r, int S, int kb, int hf) {
  const v8h lo = *(const v8h*)(p + (size_t)r * S + kb + hf * 16);
  const v8h hi = *(const v8h*)(p + (size_t)r * S + kb + hf * 16 + 8);
  v16h a;
  #pragma unroll
  for (int i = 0; i < 8; i++) { a[i] = lo[i]; a[i + 8] = hi[i]; }
  return a;
}

// Same A fragment sourced from an f32 matrix, branchless pad->0 (clamped index + select)
__device__ __forceinline__ v16h load_a32_guard(const float* p, int r, int S, int d, int kb, int hf) {
  const float* row = p + (size_t)r * S;
  v16h a;
  #pragma unroll
  for (int i = 0; i < 8; i++) {
    const int k0 = kb + hf * 16 + i;
    const int k1 = kb + hf * 16 + 8 + i;
    const float f0 = row[k0 < d ? k0 : d - 1];
    const float f1 = row[k1 < d ? k1 : d - 1];
    a[i]     = (_Float16)(k0 < d ? f0 : 0.0f);
    a[i + 8] = (_Float16)(k1 < d ? f1 : 0.0f);
  }
  return a;
}

// ---- pack weights: W16[z][n*PAD + k] = W[n*257 + k], zero padded ----
__global__ __launch_bounds__(256) void pack_w(const float* Wq, const float* Wk,
                                              const float* Wv, const float* Wo,
                                              _Float16* W16) {
  const float* src = (blockIdx.y == 0) ? Wq : (blockIdx.y == 1) ? Wk
                   : (blockIdx.y == 2) ? Wv : Wo;
  _Float16* dst = W16 + (size_t)blockIdx.y * PAD * PAD;
  const int idx = blockIdx.x * 256 + threadIdx.x;       // < PAD*PAD
  const int n = idx / PAD, k = idx % PAD;
  dst[idx] = (n < ATT && k < ATT) ? (_Float16)src[n * ATT + k] : (_Float16)0.0f;
}

__global__ __launch_bounds__(288) void pack_b(const float* bq, const float* bk,
                                              const float* bv, const float* bo,
                                              float* bp) {
  const float* src = (blockIdx.y == 0) ? bq : (blockIdx.y == 1) ? bk
                   : (blockIdx.y == 2) ? bv : bo;
  const int n = threadIdx.x;
  bp[blockIdx.y * PAD + n] = (n < ATT) ? src[n] : 0.0f;
}

// ---- QKV projection: one wave per 16-row tile; q,k -> row-major f16; v -> transposed ----
__global__ __launch_bounds__(32) void proj_qkv(
    const float* __restrict__ q, const float* __restrict__ k, const float* __restrict__ v,
    const _Float16* __restrict__ W16, const float* __restrict__ bp,
    _Float16* __restrict__ qp, _Float16* __restrict__ kp, _Float16* __restrict__ vpT) {
  const int z = blockIdx.y;
  const float* x = (z == 0) ? q : (z == 1) ? k : v;
  const _Float16* W = W16 + (size_t)z * PAD * PAD;
  const float* bias = bp + z * PAD;
  const int t0 = blockIdx.x * 16;
  const int l = threadIdx.x, l15 = l & 15, hf = l >> 4;

  v16h af[KSTEPS];
  #pragma unroll
  for (int ks = 0; ks < KSTEPS; ks++)
    af[ks] = load_a32_guard(x, t0 + l15, AMP, AMP, ks * 32, hf);

  // Uniform destination selection hoisted out of the hot loop.
  _Float16* dstRM = (z == 0) ? qp : kp;                       // row-major targets
  const int bb2 = t0 >> 11, tbase = t0 & 2047;                // for transposed v
  _Float16* dstT = vpT + (size_t)bb2 * PAD * TSEQ;

  for (int n = 0; n < NT; n++) {
    const int col = n * 16 + l15;
    v8f acc = {};
    #pragma unroll
    for (int ks = 0; ks < KSTEPS; ks++) {
      // B[K][N] with N = lane's column: contiguous 32B per lane (W stored [N][K])
      v16h bf = *(const v16h*)(W + (size_t)col * PAD + ks * 32 + hf * 16);
      acc = wmma_f16(af[ks], bf, acc);
    }
    const float bb = bias[col];
    if (z == 2) {
      #pragma unroll
      for (int vv = 0; vv < 8; vv++) {
        const int tt = tbase + vv + hf * 8;
        dstT[(size_t)col * TSEQ + tt] = (_Float16)(acc[vv] + bb);
      }
    } else {
      #pragma unroll
      for (int vv = 0; vv < 8; vv++) {
        const int row = t0 + vv + hf * 8;
        dstRM[(size_t)row * PAD + col] = (_Float16)(acc[vv] + bb);
      }
    }
  }
}

// ---- fused sigmoid attention: ctx = sum_s sigmoid(qp.kp_s * scale) * vp_s ----
// 2 waves per block share one 16x32 score tile: wave w computes score sub-tile
// st = w (9 WMMAs), both exchange via LDS + workgroup split-barrier, then each
// wave consumes the full K=32 score A-fragment for its own 9 e-tiles.
__global__ __launch_bounds__(64) void attn_fused(
    const _Float16* __restrict__ qp, const _Float16* __restrict__ kp,
    const _Float16* __restrict__ vpT, _Float16* __restrict__ ctx16) {
  const int t0 = blockIdx.x * 16;
  const int b  = t0 >> 11;
  const _Float16* kpb = kp  + (size_t)b * TSEQ * PAD;
  const _Float16* vpb = vpT + (size_t)b * PAD * TSEQ;
  const int w = threadIdx.x >> 5;             // wave id: 0/1 -> score st, e-half
  const int l = threadIdx.x & 31, l15 = l & 15, hf = l >> 4;
  const float scale = 0.062378286f;           // 1/sqrt(257)

  __shared__ _Float16 sc[16 * 32];            // score tile staging (C/D -> A relayout)

  v16h aq[KSTEPS];
  #pragma unroll
  for (int ks = 0; ks < KSTEPS; ks++)
    aq[ks] = load_a16(qp, t0 + l15, PAD, ks * 32, hf);

  const v8f vzero = {};
  v8f ctxa[9];
  #pragma unroll
  for (int n = 0; n < 9; n++) ctxa[n] = vzero;

  for (int s0 = 0; s0 < TSEQ; s0 += 32) {
    // --- this wave's 16x16 score tile: s columns [s0 + w*16, s0 + w*16 + 16) ---
    {
      const int srow = s0 + w * 16 + l15;     // B column N = kp row s: contiguous 32B loads
      v8f acc = vzero;
      #pragma unroll
      for (int ks = 0; ks < KSTEPS; ks++) {
        v16h bf = *(const v16h*)(kpb + (size_t)srow * PAD + ks * 32 + hf * 16);
        acc = wmma_f16(aq[ks], bf, acc);
      }
      #pragma unroll
      for (int vv = 0; vv < 8; vv++) {
        const float s = 1.0f / (1.0f + __expf(-acc[vv] * scale));
        sc[(vv + hf * 8) * 32 + w * 16 + l15] = (_Float16)s;  // [m][s] tile
      }
    }
    __syncthreads();                          // score tile visible to both waves
    v16h as;
    {
      const v8h lo = *(const v8h*)(sc + l15 * 32 + hf * 16);
      const v8h hi = *(const v8h*)(sc + l15 * 32 + hf * 16 + 8);
      #pragma unroll
      for (int i = 0; i < 8; i++) { as[i] = lo[i]; as[i + 8] = hi[i]; }
    }
    if (s0 + 32 < TSEQ)                       // -> global_prefetch_b8 of next kp chunk
      __builtin_prefetch(kpb + (size_t)(s0 + 32 + l) * PAD, 0, 1);
    #pragma unroll
    for (int n = 0; n < 9; n++) {
      const int e = (w * 9 + n) * 16 + l15;   // vp transposed: contiguous 32B per lane
      v16h bf = *(const v16h*)(vpb + (size_t)e * TSEQ + s0 + hf * 16);
      ctxa[n] = wmma_f16(as, bf, ctxa[n]);
    }
    __syncthreads();                          // protect LDS WAR before next chunk's writes
  }

  #pragma unroll
  for (int n = 0; n < 9; n++) {
    const int col = (w * 9 + n) * 16 + l15;
    #pragma unroll
    for (int vv = 0; vv < 8; vv++) {
      const int row = t0 + vv + hf * 8;
      ctx16[(size_t)row * PAD + col] = (_Float16)ctxa[n][vv];
    }
  }
}

// ---- output projection: out = ctx @ Wo^T + bo (f32 result) ----
__global__ __launch_bounds__(32) void out_proj(
    const _Float16* __restrict__ ctx16, const _Float16* __restrict__ Wo16,
    const float* __restrict__ bop, float* __restrict__ out) {
  const int t0 = blockIdx.x * 16;
  const int l = threadIdx.x, l15 = l & 15, hf = l >> 4;
  v16h af[KSTEPS];
  #pragma unroll
  for (int ks = 0; ks < KSTEPS; ks++)
    af[ks] = load_a16(ctx16, t0 + l15, PAD, ks * 32, hf);

  for (int n = 0; n < NT; n++) {
    const int col = n * 16 + l15;
    v8f acc = {};
    #pragma unroll
    for (int ks = 0; ks < KSTEPS; ks++) {
      v16h bf = *(const v16h*)(Wo16 + (size_t)col * PAD + ks * 32 + hf * 16);
      acc = wmma_f16(af[ks], bf, acc);
    }
    if (col < AMP) {
      const float bb = bop[col];
      #pragma unroll
      for (int vv = 0; vv < 8; vv++) {
        const int row = t0 + vv + hf * 8;
        out[(size_t)row * AMP + col] = acc[vv] + bb;
      }
    }
  }
}

extern "C" void kernel_launch(void* const* d_in, const int* in_sizes, int n_in,
                              void* d_out, int out_size, void* d_ws, size_t ws_size,
                              hipStream_t stream) {
  (void)in_sizes; (void)n_in; (void)out_size; (void)ws_size;
  const float* q  = (const float*)d_in[0];
  const float* k  = (const float*)d_in[1];
  const float* v  = (const float*)d_in[2];
  const float* Wq = (const float*)d_in[3];
  const float* bq = (const float*)d_in[4];
  const float* Wk = (const float*)d_in[5];
  const float* bk = (const float*)d_in[6];
  const float* Wv = (const float*)d_in[7];
  const float* bv = (const float*)d_in[8];
  const float* Wo = (const float*)d_in[9];
  const float* bo = (const float*)d_in[10];
  float* out = (float*)d_out;

  char* w = (char*)d_ws;
  const size_t SZ_ACT = (size_t)ROWS * PAD * sizeof(_Float16);   // 9,437,184 B each
  _Float16* qp16  = (_Float16*)(w + 0 * SZ_ACT);
  _Float16* kp16  = (_Float16*)(w + 1 * SZ_ACT);
  _Float16* vpT16 = (_Float16*)(w + 2 * SZ_ACT);
  _Float16* ctx16 = (_Float16*)(w + 3 * SZ_ACT);
  _Float16* W16   = (_Float16*)(w + 4 * SZ_ACT);                 // 4 x 288x288 f16
  float*    bp    = (float*)(w + 4 * SZ_ACT + 4 * (size_t)PAD * PAD * sizeof(_Float16));

  pack_w<<<dim3(PAD * PAD / 256, 4), 256, 0, stream>>>(Wq, Wk, Wv, Wo, W16);
  pack_b<<<dim3(1, 4), PAD, 0, stream>>>(bq, bk, bv, bo, bp);
  proj_qkv<<<dim3(RTILES, 3), 32, 0, stream>>>(q, k, v, W16, bp, qp16, kp16, vpT16);
  attn_fused<<<dim3(RTILES), 64, 0, stream>>>(qp16, kp16, vpT16, ctx16);
  out_proj<<<dim3(RTILES), 32, 0, stream>>>(ctx16, W16 + 3 * (size_t)PAD * PAD, bp + 3 * PAD, out);
}
